// MilliesRNN_89601607729294
// MI455X (gfx1250) — matrix operand order repaired and checked by reading, not compiled, not run on hardware
//
#include <hip/hip_runtime.h>
#include <hip/hip_bf16.h>

// ---------------------------------------------------------------------------
// MilliesRNN on MI455X (gfx1250): bf16 WMMA everywhere, persistent cooperative
// scan kernels with LDS-resident recurrent weights staged by the Tensor Data
// Mover (TDM) and completion tracked with TENSORcnt.
// ---------------------------------------------------------------------------

typedef __attribute__((ext_vector_type(16))) __bf16 v16bf;
typedef __attribute__((ext_vector_type(8)))  __bf16 v8bf;
typedef __attribute__((ext_vector_type(8)))  float  v8f;
typedef __attribute__((ext_vector_type(4)))  unsigned v4u;
typedef __attribute__((ext_vector_type(8)))  int    v8i;
typedef __attribute__((ext_vector_type(4)))  int    v4i;

#define DEV __device__ __forceinline__

// Problem constants
#define NB 64      // batch
#define TT 512     // time steps
#define II 512     // input dim
#define HH 1024    // hidden dim
#define OO 512     // output dim
#define MROWS (NB * TT)   // 32768 rows in all batched GEMMs

#if defined(__gfx1250__) && __has_builtin(__builtin_amdgcn_tensor_load_to_lds)
#define USE_TDM 1
#endif

// ---------------------------------------------------------------------------
// WMMA fragment helpers (contiguous k = 16*(L/16) + s for both A and B, wave32)
// ---------------------------------------------------------------------------

// A fragment: 16x32 bf16. Lane L: M=L%16, K = (L/16)*16 .. +15 (contiguous).
DEV v16bf load_a_frag(const __bf16* A, int lda, int row0, int k0, int lane) {
  const int m = lane & 15, half = lane >> 4;
  return *(const v16bf*)(A + (size_t)(row0 + m) * lda + k0 + half * 16);
}

// B fragment: 32x16 bf16, B[k][n] = W[n][k] (weights stored (out,in) row-major).
// Lane L: N=L%16, K = (L/16)*16 .. +15 -> one contiguous 32B read of W row.
DEV v16bf load_b_frag(const __bf16* W, int ldw, int n0, int k0, int lane) {
  const int n = lane & 15, half = lane >> 4;
  return *(const v16bf*)(W + (size_t)(n0 + n) * ldw + k0 + half * 16);
}

DEV v8f wmma_bf16(v16bf a, v16bf b, v8f c) {
  return __builtin_amdgcn_wmma_f32_16x16x32_bf16(false, a, false, b,
                                                 (short)0, c, false, false);
}

DEV float retanh(float x) { return tanhf(fmaxf(x, 0.0f)); }

// ---------------------------------------------------------------------------
// TDM: stage a [rows x cols] bf16 tile (row-major, stride == cols) from global
// memory into LDS at byte offset ldsOff. Descriptor per CDNA5 ISA ch.8:
//   group0: count=1 | lds_addr | global_addr[56:0] | type=2
//   group1: data_size=1(2B) | tensor_dim0=cols | tensor_dim1=rows |
//           tile_dim0=cols | tile_dim1=rows | tensor_dim0_stride=cols
//   groups 2/3: zero (<=2D tensor)
// Issued by one wave; completion = s_wait_tensorcnt 0.
// ---------------------------------------------------------------------------
#ifdef USE_TDM
DEV void tdm_load_tile_to_lds(const __bf16* gsrc, unsigned ldsOff,
                              unsigned rows, unsigned cols) {
  unsigned long long ga = (unsigned long long)(__UINTPTR_TYPE__)gsrc;
  v4u g0;
  g0[0] = 1u;                                    // count=1, user descriptor
  g0[1] = ldsOff;                                // lds_addr (bytes)
  g0[2] = (unsigned)(ga & 0xFFFFFFFFu);          // global_addr[31:0]
  g0[3] = (unsigned)((ga >> 32) & 0x01FFFFFFu)   // global_addr[56:32]
        | (2u << 30);                            // type=2 ("image")
  v8i g1 = {};
  g1[0] = (int)(1u << 16);                       // wg_mask=0, data_size=1 (2B)
  g1[1] = (int)((cols & 0xFFFFu) << 16);         // tensor_dim0[15:0]
  g1[2] = (int)((cols >> 16) | ((rows & 0xFFFFu) << 16));  // dim0 hi | dim1 lo
  g1[3] = (int)((rows >> 16) | ((cols & 0xFFFFu) << 16));  // dim1 hi | tile_dim0
  g1[4] = (int)rows;                             // tile_dim1 (tile_dim2 = 0)
  g1[5] = (int)cols;                             // tensor_dim0_stride[31:0]
  v4i gz = {};
#if __clang_major__ >= 23
  __builtin_amdgcn_tensor_load_to_lds(g0, g1, gz, gz, (v8i){}, 0);
#else
  __builtin_amdgcn_tensor_load_to_lds(g0, g1, gz, gz, 0);
#endif
  __builtin_amdgcn_s_wait_tensorcnt(0);
}
#endif

// ---------------------------------------------------------------------------
// Device-wide barrier (persistent cooperative scan kernels)
// ---------------------------------------------------------------------------
DEV void grid_sync(unsigned* cnt, unsigned* gen, unsigned nblocks) {
  __threadfence();
  __syncthreads();
  if (threadIdx.x == 0) {
    unsigned old = __hip_atomic_load(gen, __ATOMIC_RELAXED, __HIP_MEMORY_SCOPE_AGENT);
    unsigned arrived = __hip_atomic_fetch_add(cnt, 1u, __ATOMIC_ACQ_REL,
                                              __HIP_MEMORY_SCOPE_AGENT);
    if (arrived == nblocks - 1) {
      __hip_atomic_store(cnt, 0u, __ATOMIC_RELAXED, __HIP_MEMORY_SCOPE_AGENT);
      __hip_atomic_fetch_add(gen, 1u, __ATOMIC_ACQ_REL, __HIP_MEMORY_SCOPE_AGENT);
    } else {
      while (__hip_atomic_load(gen, __ATOMIC_ACQUIRE, __HIP_MEMORY_SCOPE_AGENT) == old) {
        __builtin_amdgcn_s_sleep(2);
      }
    }
  }
  __syncthreads();
}

// ---------------------------------------------------------------------------
// fp32 -> bf16 conversion
// ---------------------------------------------------------------------------
__global__ void cvt_f32_bf16(const float* __restrict__ in, __bf16* __restrict__ out,
                             int n) {
  for (int i = blockIdx.x * 256 + threadIdx.x; i < n; i += gridDim.x * 256)
    out[i] = (__bf16)in[i];
}

// ---------------------------------------------------------------------------
// Generic bf16 WMMA GEMM:  out[r][c] = act( A[r][:] . W[c][:] + bias[c] )
// Block = 256 threads = 8 waves; block tile 64(M) x 128(N);
// wave w: m-tile = w&3, n-tiles = (w>>2)*64 .. +63 (4 tiles of 16).
// PERM maps row r = t*NB+n -> output row n*TT+t (for the final [n,t,o] store).
// ---------------------------------------------------------------------------
template <bool RETANH, bool OUT_BF16, bool PERM>
__global__ void gemm_bf16_kernel(const __bf16* __restrict__ A, int lda,
                                 const __bf16* __restrict__ W, int ldw,
                                 const float* __restrict__ bias,
                                 float* __restrict__ outF,
                                 __bf16* __restrict__ outB, int ldo,
                                 int K) {
  const int tid = threadIdx.x;
  const int lane = tid & 31, wave = tid >> 5;
  const int row0 = blockIdx.x * 64 + (wave & 3) * 16;
  const int n0   = blockIdx.y * 128 + (wave >> 2) * 64;

  v8f acc[4] = {};
  for (int k = 0; k < K; k += 32) {
    v16bf a = load_a_frag(A, lda, row0, k, lane);
#pragma unroll
    for (int j = 0; j < 4; ++j) {
      v16bf b = load_b_frag(W, ldw, n0 + j * 16, k, lane);
      acc[j] = wmma_bf16(a, b, acc[j]);
    }
  }

  const int nl = lane & 15, mh = lane >> 4;
#pragma unroll
  for (int j = 0; j < 4; ++j) {
    const int col = n0 + j * 16 + nl;
    const float bv = bias[col];
#pragma unroll
    for (int v = 0; v < 8; ++v) {
      const int r = row0 + v + 8 * mh;
      float x = acc[j][v] + bv;
      if (RETANH) x = retanh(x);
      size_t orow = PERM ? ((size_t)(r & (NB - 1)) * TT + (size_t)(r >> 6))
                         : (size_t)r;
      if (OUT_BF16) outB[orow * ldo + col] = (__bf16)x;
      else          outF[orow * ldo + col] = x;
    }
  }
}

// ---------------------------------------------------------------------------
// Persistent RNN scan:  h_{t} = retanh(inp_t + h_{t-1} @ Wh^T + bh)
// 16 blocks, each owns 64 columns of H; its 64x1024 bf16 Wh slice (128 KB)
// is DMA'd once into LDS by the TDM (320 KB/WGP on CDNA5). Hidden state
// ping-pongs between two global bf16 buffers with one device-wide barrier
// per step. inp element (n, t, col) read at n*strideN + t*strideT + col.
// hs written bf16 at (t*NB + n)*HH + col.
// ---------------------------------------------------------------------------
__global__ void rnn_scan_kernel(const float* __restrict__ inp,
                                long strideN, long strideT,
                                const float* __restrict__ h0,
                                const __bf16* __restrict__ Wh,   // [HH, HH] bf16
                                const float* __restrict__ bh,
                                __bf16* __restrict__ hs,         // [TT*NB, HH]
                                __bf16* __restrict__ hst0,
                                __bf16* __restrict__ hst1,
                                unsigned* __restrict__ bar,
                                unsigned nblocks) {
  extern __shared__ __align__(32) char smem_raw[];
  __bf16* WhS = (__bf16*)smem_raw;                  // [64][HH] local slice

  const int tid = threadIdx.x;
  const int lane = tid & 31, wave = tid >> 5;
  const int colBase = blockIdx.x * 64;

  // Stage this block's Wh slice (rows colBase..colBase+63) into LDS.
#ifdef USE_TDM
  if (wave == 0)
    tdm_load_tile_to_lds(Wh + (size_t)colBase * HH, 0u, 64u, (unsigned)HH);
  __syncthreads();   // LDS visible to all waves after TENSORcnt==0 + barrier
#else
  {
    const int total = 64 * HH;
    const __bf16* src = Wh + (size_t)colBase * HH;
    for (int i = tid * 8; i < total; i += 256 * 8)
      *(v8bf*)&WhS[i] = *(const v8bf*)&src[i];
  }
  __syncthreads();
#endif

  // Initialize hidden state (this block's column slice) from fp32 h0.
  for (int i = tid; i < NB * 64; i += 256) {
    const int n = i >> 6, c = colBase + (i & 63);
    hst0[(size_t)n * HH + c] = (__bf16)h0[(size_t)n * HH + c];
  }
  grid_sync(bar, bar + 1, nblocks);

  // Wave tiling: m-tile = wave&3 (batch rows), n-tiles = (wave>>2)*2 + {0,1}.
  const int mt = wave & 3;
  const int grp = wave >> 2;
  const int row0 = mt * 16;
  const int nl = lane & 15, mh = lane >> 4;

  for (int t = 0; t < TT; ++t) {
    const __bf16* hr = (t & 1) ? hst1 : hst0;   // read state
    __bf16*       hw = (t & 1) ? hst0 : hst1;   // write next state

    v8f acc0 = {}, acc1 = {};
    for (int k = 0; k < HH; k += 32) {
      v16bf a  = load_a_frag(hr, HH, row0, k, lane);
      v16bf b0 = load_b_frag(WhS, HH, (grp * 2 + 0) * 16, k, lane);
      v16bf b1 = load_b_frag(WhS, HH, (grp * 2 + 1) * 16, k, lane);
      acc0 = wmma_bf16(a, b0, acc0);
      acc1 = wmma_bf16(a, b1, acc1);
    }

#pragma unroll
    for (int j = 0; j < 2; ++j) {
      v8f acc = j ? acc1 : acc0;
      const int col = colBase + (grp * 2 + j) * 16 + nl;
      const float bv = bh[col];
#pragma unroll
      for (int v = 0; v < 8; ++v) {
        const int n = row0 + v + 8 * mh;   // batch index
        float x = acc[v] + inp[(size_t)n * strideN + (size_t)t * strideT + col] + bv;
        float y = retanh(x);
        __bf16 yb = (__bf16)y;
        hs[((size_t)t * NB + n) * HH + col] = yb;
        hw[(size_t)n * HH + col] = yb;
      }
    }
    grid_sync(bar, bar + 1, nblocks);
  }
}

// ---------------------------------------------------------------------------
// Host-side orchestration
// ---------------------------------------------------------------------------
extern "C" void kernel_launch(void* const* d_in, const int* in_sizes, int n_in,
                              void* d_out, int out_size, void* d_ws, size_t ws_size,
                              hipStream_t stream) {
  const float* data = (const float*)d_in[0];
  const float* h0_v = (const float*)d_in[1];
  const float* h0_m = (const float*)d_in[2];
  const float* Wi   = (const float*)d_in[3];
  const float* bi   = (const float*)d_in[4];
  const float* Wh   = (const float*)d_in[5];
  const float* bh   = (const float*)d_in[6];
  const float* Wo   = (const float*)d_in[7];
  const float* bo   = (const float*)d_in[8];
  const float* Wt   = (const float*)d_in[9];
  const float* bt   = (const float*)d_in[10];
  const float* Wi2  = (const float*)d_in[11];
  const float* bi2  = (const float*)d_in[12];
  const float* Wh2  = (const float*)d_in[13];
  const float* bh2  = (const float*)d_in[14];
  const float* Wo2  = (const float*)d_in[15];
  const float* bo2  = (const float*)d_in[16];
  float* out = (float*)d_out;

  // Bump allocator over workspace (256B aligned chunks).
  char* ws = (char*)d_ws;
  size_t off = 0;
  auto alloc = [&](size_t bytes) -> char* {
    char* p = ws + off;
    off += (bytes + 255) & ~(size_t)255;
    return p;
  };

  unsigned* bar   = (unsigned*)alloc(256);
  __bf16* dataB   = (__bf16*)alloc((size_t)MROWS * II * 2);
  __bf16* WiB     = (__bf16*)alloc((size_t)HH * II * 2);
  __bf16* WhB     = (__bf16*)alloc((size_t)HH * HH * 2);
  __bf16* WoB     = (__bf16*)alloc((size_t)OO * HH * 2);
  __bf16* WtB     = (__bf16*)alloc((size_t)OO * OO * 2);
  __bf16* Wi2B    = (__bf16*)alloc((size_t)HH * OO * 2);
  __bf16* Wh2B    = (__bf16*)alloc((size_t)HH * HH * 2);
  __bf16* Wo2B    = (__bf16*)alloc((size_t)OO * HH * 2);
  __bf16* hst0    = (__bf16*)alloc((size_t)NB * HH * 2);
  __bf16* hst1    = (__bf16*)alloc((size_t)NB * HH * 2);
  float*  inpF    = (float*) alloc((size_t)MROWS * HH * 4);   // input projections
  __bf16* hsB     = (__bf16*)alloc((size_t)MROWS * HH * 2);   // hidden states (reused)
  __bf16* outvB   = (__bf16*)alloc((size_t)MROWS * OO * 2);
  __bf16* outtB   = (__bf16*)alloc((size_t)MROWS * OO * 2);
  (void)ws_size; (void)in_sizes; (void)n_in; (void)out_size;

  hipMemsetAsync(bar, 0, 256, stream);

  // ---- fp32 -> bf16 conversions -------------------------------------------
  auto cvt = [&](const float* src, __bf16* dst, int n) {
    int blocks = (n + 255) / 256; if (blocks > 4096) blocks = 4096;
    cvt_f32_bf16<<<blocks, 256, 0, stream>>>(src, dst, n);
  };
  cvt(data, dataB, MROWS * II);
  cvt(Wi,  WiB,  HH * II);
  cvt(Wh,  WhB,  HH * HH);
  cvt(Wo,  WoB,  OO * HH);
  cvt(Wt,  WtB,  OO * OO);
  cvt(Wi2, Wi2B, HH * OO);
  cvt(Wh2, Wh2B, HH * HH);
  cvt(Wo2, Wo2B, OO * HH);

  const dim3 blk(256);
  const size_t scanLds = (size_t)64 * HH * sizeof(__bf16);   // 128 KB / block

  // ---- 1) inp_v = data @ Wi^T + bi  (rows r = n*T+t, fp32 out) ------------
  gemm_bf16_kernel<false, false, false><<<dim3(MROWS / 64, HH / 128), blk, 0, stream>>>(
      dataB, II, WiB, II, bi, inpF, nullptr, HH, II);

  // ---- 2) visual scan (inp rows n*T+t: strideN = T*H, strideT = H) --------
  rnn_scan_kernel<<<dim3(HH / 64), blk, scanLds, stream>>>(
      inpF, (long)TT * HH, (long)HH, h0_v, WhB, bh, hsB, hst0, hst1, bar, HH / 64);

  // ---- 3) out_v = hs_v @ Wo^T + bo  (bf16 out) ----------------------------
  gemm_bf16_kernel<false, true, false><<<dim3(MROWS / 64, OO / 128), blk, 0, stream>>>(
      hsB, HH, WoB, HH, bo, nullptr, outvB, OO, HH);

  // ---- 4) out_t = retanh(out_v @ Wt^T + bt)  (bf16 out) -------------------
  gemm_bf16_kernel<true, true, false><<<dim3(MROWS / 64, OO / 128), blk, 0, stream>>>(
      outvB, OO, WtB, OO, bt, nullptr, outtB, OO, OO);

  // ---- 5) inp_m = out_t @ Wi2^T + bi2  (rows r = t*NB+n, fp32 out) --------
  gemm_bf16_kernel<false, false, false><<<dim3(MROWS / 64, HH / 128), blk, 0, stream>>>(
      outtB, OO, Wi2B, OO, bi2, inpF, nullptr, HH, OO);

  // ---- 6) motor scan (inp rows t*NB+n: strideN = H, strideT = NB*H) -------
  rnn_scan_kernel<<<dim3(HH / 64), blk, scanLds, stream>>>(
      inpF, (long)HH, (long)NB * HH, h0_m, Wh2B, bh2, hsB, hst0, hst1, bar, HH / 64);

  // ---- 7) out_m = hs_m @ Wo2^T + bo2, stored [n,t,o] (row permute) --------
  gemm_bf16_kernel<false, false, true><<<dim3(MROWS / 64, OO / 128), blk, 0, stream>>>(
      hsB, HH, Wo2B, HH, bo2, out, nullptr, OO, HH);
}
